// RingAttention_54142357733523
// MI455X (gfx1250) — matrix operand order repaired and checked by reading, not compiled, not run on hardware
//
#include <hip/hip_runtime.h>
#include <hip/hip_bf16.h>

typedef __attribute__((ext_vector_type(16))) _Float16 v16h;
typedef __attribute__((ext_vector_type(8)))  _Float16 v8h;
typedef __attribute__((ext_vector_type(8)))  float    v8f;

union V16H { v16h v; v8h h2[2]; _Float16 e[16]; };

#define WMMA_F16(A, B, C) \
  __builtin_amdgcn_wmma_f32_16x16x32_f16(false, (A), false, (B), (short)0, (C), false, false)

// ---- CDNA5 async global->LDS copy (ASYNCcnt-tracked), per CDNA5 ISA ----
__device__ __forceinline__ unsigned lds_off(const void* p) {
  // generic pointer to LDS: addr[31:0] is the wave-relative LDS byte address
  return (unsigned)(uintptr_t)p;
}
__device__ __forceinline__ void async_b128(unsigned lds, unsigned long long gaddr) {
  asm volatile("global_load_async_to_lds_b128 %0, %1, off"
               :: "v"(lds), "v"(gaddr) : "memory");
}
__device__ __forceinline__ void wait_async0() {
  asm volatile("s_wait_asynccnt 0x0" ::: "memory");
}

// ---------------------------------------------------------------------------
// Kernel 1: LayerNorm + convert to f16.  One row (1024 elems) per 256-thread block.
// ---------------------------------------------------------------------------
__global__ __launch_bounds__(256)
void ln_f16_kernel(const float* __restrict__ x, const float* __restrict__ gamma,
                   const float* __restrict__ beta, _Float16* __restrict__ xn) {
  __shared__ float red[256];
  const int t = threadIdx.x;
  const size_t row = blockIdx.x;
  const float* xr = x + row * 1024;

  float4 v4 = *(const float4*)(xr + t * 4);
  red[t] = v4.x + v4.y + v4.z + v4.w; __syncthreads();
  for (int off = 128; off > 0; off >>= 1) {
    if (t < off) red[t] += red[t + off];
    __syncthreads();
  }
  const float mu = red[0] * (1.0f / 1024.0f);
  __syncthreads();

  float d0 = v4.x - mu, d1 = v4.y - mu, d2 = v4.z - mu, d3 = v4.w - mu;
  red[t] = d0*d0 + d1*d1 + d2*d2 + d3*d3; __syncthreads();
  for (int off = 128; off > 0; off >>= 1) {
    if (t < off) red[t] += red[t + off];
    __syncthreads();
  }
  const float rs = rsqrtf(red[0] * (1.0f / 1024.0f) + 1e-5f);

  _Float16* xo = xn + row * 1024 + t * 4;
  const float* g = gamma + t * 4;
  const float* b = beta  + t * 4;
  xo[0] = (_Float16)(d0 * rs * g[0] + b[0]);
  xo[1] = (_Float16)(d1 * rs * g[1] + b[1]);
  xo[2] = (_Float16)(d2 * rs * g[2] + b[2]);
  xo[3] = (_Float16)(d3 * rs * g[3] + b[3]);
}

// ---------------------------------------------------------------------------
// Kernel 2: transpose + convert weight  w[K][N] f32 -> wt[N][K] f16
// ---------------------------------------------------------------------------
__global__ __launch_bounds__(256)
void convT_kernel(const float* __restrict__ w, _Float16* __restrict__ wt,
                  int K, int N) {
  int idx = blockIdx.x * 256 + threadIdx.x;
  if (idx < K * N) {
    int k = idx / N, n = idx % N;
    wt[(size_t)n * K + k] = (_Float16)w[idx];
  }
}

// ---------------------------------------------------------------------------
// Kernel 3: QKV GEMM.  C[8192][1536] = Xn[8192][1024] @ Wqkv[1024][1536]
// Block tile 128x64, 8 waves (4 in M x 2 in N), each wave 32x32 = 4 WMMAs/K-step.
// Async global->LDS staging, double-buffered.  Epilogue scatters q/k/v;
// q scaled by dh^-0.5; v written TRANSPOSED [b][h][dh][n] for attention.
// ---------------------------------------------------------------------------
__global__ __launch_bounds__(256)
void gemm_qkv_kernel(const _Float16* __restrict__ A, const _Float16* __restrict__ Bt,
                     _Float16* __restrict__ qo, _Float16* __restrict__ ko,
                     _Float16* __restrict__ vt) {
  __shared__ __align__(64) _Float16 lA[2][128 * 32];
  __shared__ __align__(64) _Float16 lB[2][64 * 32];
  const int t = threadIdx.x, lane = t & 31, w = t >> 5;
  const int wm = w & 3, wn = w >> 2;
  const int bm = blockIdx.y * 128, bn = blockIdx.x * 64;
  const int K = 1024;

  const int arow = t >> 1, aoff = (t & 1) * 32;   // A: 128 rows x 64B, 32B/thread
  const int brow = t >> 2, boff = (t & 3) * 16;   // B: 64 rows x 64B, 16B/thread
  const unsigned long long gA = (unsigned long long)(uintptr_t)A;
  const unsigned long long gB = (unsigned long long)(uintptr_t)Bt;

  v8f a00 = {}, a01 = {}, a10 = {}, a11 = {};
  const int fr0  = 32 * wm + (lane & 15);
  const int c0   = (lane < 16) ? 0 : 16;
  const int koff = (lane < 16) ? 0 : 16;
  const int br0  = 32 * wn + (lane & 15);

#define ISSUE_TILES(buf, kb)                                                    \
  {                                                                             \
    unsigned la = lds_off(&lA[buf][0]) + (unsigned)(arow * 64 + aoff);          \
    unsigned long long ga = gA + ((size_t)(bm + arow) * K + (kb)) * 2 + aoff;   \
    async_b128(la, ga); async_b128(la + 16, ga + 16);                           \
    unsigned lb = lds_off(&lB[buf][0]) + (unsigned)(brow * 64 + boff);          \
    unsigned long long gb = gB + ((size_t)(bn + brow) * K + (kb)) * 2 + boff;   \
    async_b128(lb, gb);                                                         \
  }

  ISSUE_TILES(0, 0);
  wait_async0();
  __syncthreads();

  int cur = 0;
  for (int kb = 0; kb < K; kb += 32) {
    const bool more = (kb + 32) < K;
    if (more) ISSUE_TILES(cur ^ 1, kb + 32);

    const _Float16* pa = &lA[cur][0];
    const _Float16* pb = &lB[cur][0];
    V16H af0, af1;
    af0.h2[0] = *(const v8h*)(pa + fr0 * 32 + c0);
    af0.h2[1] = *(const v8h*)(pa + fr0 * 32 + c0 + 8);
    af1.h2[0] = *(const v8h*)(pa + (fr0 + 16) * 32 + c0);
    af1.h2[1] = *(const v8h*)(pa + (fr0 + 16) * 32 + c0 + 8);
    v16h bf0 = *(const v16h*)(pb + br0 * 32 + koff);
    v16h bf1 = *(const v16h*)(pb + (br0 + 16) * 32 + koff);
    a00 = WMMA_F16(af0.v, bf0, a00);
    a01 = WMMA_F16(af0.v, bf1, a01);
    a10 = WMMA_F16(af1.v, bf0, a10);
    a11 = WMMA_F16(af1.v, bf1, a11);

    if (more) wait_async0();
    __syncthreads();
    cur ^= 1;
  }
#undef ISSUE_TILES

  const int hi = (lane >= 16) ? 8 : 0;
  for (int i = 0; i < 2; ++i) {
    for (int r = 0; r < 8; ++r) {
      const int gm = bm + 32 * wm + 16 * i + r + hi;
      const int bidx = gm >> 12, nq = gm & 4095;
      for (int j = 0; j < 2; ++j) {
        const int gn = bn + 32 * wn + 16 * j + (lane & 15);
        float val = (i == 0) ? ((j == 0) ? a00[r] : a01[r])
                             : ((j == 0) ? a10[r] : a11[r]);
        const int which = gn >> 9;       // 0=q 1=k 2=v
        const int rem = gn & 511;
        const int hh = rem >> 6, d = rem & 63;
        if (which == 0) {
          qo[(((size_t)bidx * 8 + hh) * 4096 + nq) * 64 + d] = (_Float16)(val * 0.125f);
        } else if (which == 1) {
          ko[(((size_t)bidx * 8 + hh) * 4096 + nq) * 64 + d] = (_Float16)val;
        } else {                          // v stored transposed: [b][h][dh][n]
          vt[(((size_t)bidx * 8 + hh) * 64 + d) * 4096 + nq] = (_Float16)val;
        }
      }
    }
  }
}

// ---------------------------------------------------------------------------
// Kernel 4: causal flash attention.  Grid (64 q-tiles, 16 b*h), block = 4 waves.
// K tile [kv][dh] and V^T tile [dh][kv] staged via async global->LDS.
// ---------------------------------------------------------------------------
__global__ __launch_bounds__(128)
void attn_kernel(const _Float16* __restrict__ q, const _Float16* __restrict__ k,
                 const _Float16* __restrict__ vt, _Float16* __restrict__ ao) {
  __shared__ __align__(64) _Float16 lk [32 * 64];     // K tile  [kv][dh]
  __shared__ __align__(64) _Float16 lvt[64 * 32];     // V^T tile [dh][kv]
  __shared__ __align__(64) _Float16 lp [4 * 16 * 32]; // per-wave P tiles

  const int t = threadIdx.x, lane = t & 31, w = t >> 5;
  const int bh = blockIdx.y;
  const int bi = bh >> 3, hh = bh & 7;
  const int q0 = blockIdx.x * 64;
  const _Float16* qp  = q  + (size_t)bh * 4096 * 64;
  const _Float16* kp  = k  + (size_t)bh * 4096 * 64;
  const _Float16* vtp = vt + (size_t)bh * 64 * 4096;
  const unsigned long long gK = (unsigned long long)(uintptr_t)kp;
  const unsigned long long gV = (unsigned long long)(uintptr_t)vtp;

  const int c0   = (lane < 16) ? 0 : 16;
  const int koff = (lane < 16) ? 0 : 16;
  const int hi   = (lane >= 16) ? 8 : 0;

  // Preload Q A-fragments (dh chunks 0-31, 32-63) straight from global.
  const int qrow = q0 + 16 * w + (lane & 15);
  V16H qa0, qa1;
  qa0.h2[0] = *(const v8h*)(qp + (size_t)qrow * 64 + c0);
  qa0.h2[1] = *(const v8h*)(qp + (size_t)qrow * 64 + c0 + 8);
  qa1.h2[0] = *(const v8h*)(qp + (size_t)qrow * 64 + 32 + c0);
  qa1.h2[1] = *(const v8h*)(qp + (size_t)qrow * 64 + 32 + c0 + 8);

  v8f o0 = {}, o1 = {}, o2 = {}, o3 = {};
  float mrow[8], lrow[8];
  for (int r = 0; r < 8; ++r) { mrow[r] = -3.0e38f; lrow[r] = 0.0f; }

  const int nk = q0 / 32 + 2;          // causal: kv tiles up to the diagonal
  const int vrow = t >> 1, voff = (t & 1) * 32;

  for (int kt = 0; kt < nk; ++kt) {
    __syncthreads();                   // prior iteration finished reading lk/lvt
    {  // K tile: 4096 contiguous bytes; 32B per thread
      unsigned l = lds_off(lk) + (unsigned)(t * 32);
      unsigned long long g = gK + (size_t)kt * 4096 + (size_t)t * 32;
      async_b128(l, g); async_b128(l + 16, g + 16);
    }
    {  // V^T tile: 64 rows x 64B (row stride 8192B); 32B per thread
      unsigned l = lds_off(lvt) + (unsigned)(vrow * 64 + voff);
      unsigned long long g = gV + (size_t)vrow * 8192 + (size_t)kt * 64 + voff;
      async_b128(l, g); async_b128(l + 16, g + 16);
    }
    wait_async0();
    __syncthreads();

    if (kt + 1 < nk) {                 // prefetch next tiles into caches
      __builtin_prefetch((const char*)kp  + (size_t)(kt + 1) * 4096 + t * 32, 0, 0);
      __builtin_prefetch((const char*)vtp + (size_t)vrow * 8192 + (size_t)(kt + 1) * 64, 0, 0);
    }

    // ---- S = Q K^T  (two 16-wide kv tiles, two dh chunks each) ----
    const int kv0 = lane & 15;
    v16h bk0  = *(const v16h*)(lk + kv0 * 64 + koff);
    v16h bk0b = *(const v16h*)(lk + kv0 * 64 + 32 + koff);
    v16h bk1  = *(const v16h*)(lk + (kv0 + 16) * 64 + koff);
    v16h bk1b = *(const v16h*)(lk + (kv0 + 16) * 64 + 32 + koff);
    v8f s0 = {}, s1 = {};
    s0 = WMMA_F16(qa0.v, bk0,  s0);
    s0 = WMMA_F16(qa1.v, bk0b, s0);
    s1 = WMMA_F16(qa0.v, bk1,  s1);
    s1 = WMMA_F16(qa1.v, bk1b, s1);

    // ---- online softmax in C-fragment layout ----
    const bool need_mask = (kt * 32 + 31) > (q0 + 16 * w);
    for (int r = 0; r < 8; ++r) {
      float a0 = s0[r], a1 = s1[r];
      if (need_mask) {
        const int qg  = q0 + 16 * w + r + hi;
        const int kva = kt * 32 + (lane & 15);
        if (kva > qg)      a0 = -3.0e38f;
        if (kva + 16 > qg) a1 = -3.0e38f;
      }
      float tm = fmaxf(a0, a1);
      tm = fmaxf(tm, __shfl_xor(tm, 1));
      tm = fmaxf(tm, __shfl_xor(tm, 2));
      tm = fmaxf(tm, __shfl_xor(tm, 4));
      tm = fmaxf(tm, __shfl_xor(tm, 8));
      const float mn = fmaxf(mrow[r], tm);
      const float alpha = __expf(mrow[r] - mn);
      const float e0 = __expf(a0 - mn);
      const float e1 = __expf(a1 - mn);
      float rsum = e0 + e1;
      rsum += __shfl_xor(rsum, 1);
      rsum += __shfl_xor(rsum, 2);
      rsum += __shfl_xor(rsum, 4);
      rsum += __shfl_xor(rsum, 8);
      lrow[r] = lrow[r] * alpha + rsum;
      mrow[r] = mn;
      o0[r] *= alpha; o1[r] *= alpha; o2[r] *= alpha; o3[r] *= alpha;
      s0[r] = e0; s1[r] = e1;
    }

    // ---- reshape P: C-layout regs -> A-fragment via per-wave LDS ----
    _Float16* lpw = lp + w * 16 * 32;
    for (int r = 0; r < 8; ++r) {
      const int m = r + hi;
      lpw[m * 32 + (lane & 15)]      = (_Float16)s0[r];
      lpw[m * 32 + 16 + (lane & 15)] = (_Float16)s1[r];
    }
    __syncthreads();

    V16H pa;
    pa.h2[0] = *(const v8h*)(lpw + (lane & 15) * 32 + c0);
    pa.h2[1] = *(const v8h*)(lpw + (lane & 15) * 32 + c0 + 8);

    // ---- O += P @ V   (4 dh tiles of 16) ----
    v16h bv0 = *(const v16h*)(lvt + ((lane & 15) + 0)  * 32 + koff);
    v16h bv1 = *(const v16h*)(lvt + ((lane & 15) + 16) * 32 + koff);
    v16h bv2 = *(const v16h*)(lvt + ((lane & 15) + 32) * 32 + koff);
    v16h bv3 = *(const v16h*)(lvt + ((lane & 15) + 48) * 32 + koff);
    o0 = WMMA_F16(pa.v, bv0, o0);
    o1 = WMMA_F16(pa.v, bv1, o1);
    o2 = WMMA_F16(pa.v, bv2, o2);
    o3 = WMMA_F16(pa.v, bv3, o3);
  }

  // ---- finalize: O / (l + EPS) -> attn_out [b][n][h*dh] f16 ----
  for (int r = 0; r < 8; ++r) {
    const int m = r + hi;
    const int gq = q0 + 16 * w + m;
    const float dn = 1.0f / (lrow[r] + 1e-10f);
    const size_t rowoff = ((size_t)bi * 4096 + gq) * 512 + hh * 64;
    ao[rowoff + 0  + (lane & 15)] = (_Float16)(o0[r] * dn);
    ao[rowoff + 16 + (lane & 15)] = (_Float16)(o1[r] * dn);
    ao[rowoff + 32 + (lane & 15)] = (_Float16)(o2[r] * dn);
    ao[rowoff + 48 + (lane & 15)] = (_Float16)(o3[r] * dn);
  }
}

// ---------------------------------------------------------------------------
// Kernel 5: output GEMM.  out[8192][1024] f32 = attn[8192][512] @ Wout[512][1024]
// Same tiling as QKV GEMM (128x64 block, 4 WMMAs/wave/K-step, async staging).
// ---------------------------------------------------------------------------
__global__ __launch_bounds__(256)
void gemm_out_kernel(const _Float16* __restrict__ A, const _Float16* __restrict__ Bt,
                     float* __restrict__ out) {
  __shared__ __align__(64) _Float16 lA[2][128 * 32];
  __shared__ __align__(64) _Float16 lB[2][64 * 32];
  const int t = threadIdx.x, lane = t & 31, w = t >> 5;
  const int wm = w & 3, wn = w >> 2;
  const int bm = blockIdx.y * 128, bn = blockIdx.x * 64;
  const int K = 512;

  const int arow = t >> 1, aoff = (t & 1) * 32;
  const int brow = t >> 2, boff = (t & 3) * 16;
  const unsigned long long gA = (unsigned long long)(uintptr_t)A;
  const unsigned long long gB = (unsigned long long)(uintptr_t)Bt;

  v8f a00 = {}, a01 = {}, a10 = {}, a11 = {};
  const int fr0  = 32 * wm + (lane & 15);
  const int c0   = (lane < 16) ? 0 : 16;
  const int koff = (lane < 16) ? 0 : 16;
  const int br0  = 32 * wn + (lane & 15);

#define ISSUE_TILES(buf, kb)                                                    \
  {                                                                             \
    unsigned la = lds_off(&lA[buf][0]) + (unsigned)(arow * 64 + aoff);          \
    unsigned long long ga = gA + ((size_t)(bm + arow) * K + (kb)) * 2 + aoff;   \
    async_b128(la, ga); async_b128(la + 16, ga + 16);                           \
    unsigned lb = lds_off(&lB[buf][0]) + (unsigned)(brow * 64 + boff);          \
    unsigned long long gb = gB + ((size_t)(bn + brow) * K + (kb)) * 2 + boff;   \
    async_b128(lb, gb);                                                         \
  }

  ISSUE_TILES(0, 0);
  wait_async0();
  __syncthreads();

  int cur = 0;
  for (int kb = 0; kb < K; kb += 32) {
    const bool more = (kb + 32) < K;
    if (more) ISSUE_TILES(cur ^ 1, kb + 32);

    const _Float16* pa = &lA[cur][0];
    const _Float16* pb = &lB[cur][0];
    V16H af0, af1;
    af0.h2[0] = *(const v8h*)(pa + fr0 * 32 + c0);
    af0.h2[1] = *(const v8h*)(pa + fr0 * 32 + c0 + 8);
    af1.h2[0] = *(const v8h*)(pa + (fr0 + 16) * 32 + c0);
    af1.h2[1] = *(const v8h*)(pa + (fr0 + 16) * 32 + c0 + 8);
    v16h bf0 = *(const v16h*)(pb + br0 * 32 + koff);
    v16h bf1 = *(const v16h*)(pb + (br0 + 16) * 32 + koff);
    a00 = WMMA_F16(af0.v, bf0, a00);
    a01 = WMMA_F16(af0.v, bf1, a01);
    a10 = WMMA_F16(af1.v, bf0, a10);
    a11 = WMMA_F16(af1.v, bf1, a11);

    if (more) wait_async0();
    __syncthreads();
    cur ^= 1;
  }
#undef ISSUE_TILES

  const int hi = (lane >= 16) ? 8 : 0;
  for (int i = 0; i < 2; ++i) {
    for (int r = 0; r < 8; ++r) {
      const int gm = bm + 32 * wm + 16 * i + r + hi;
      const int gn0 = bn + 32 * wn + (lane & 15);
      out[(size_t)gm * 1024 + gn0]      = (i == 0) ? a00[r] : a10[r];
      out[(size_t)gm * 1024 + gn0 + 16] = (i == 0) ? a01[r] : a11[r];
    }
  }
}

// ---------------------------------------------------------------------------
// Host-side launcher
// ---------------------------------------------------------------------------
extern "C" void kernel_launch(void* const* d_in, const int* in_sizes, int n_in,
                              void* d_out, int out_size, void* d_ws, size_t ws_size,
                              hipStream_t stream) {
  (void)in_sizes; (void)n_in; (void)out_size; (void)ws_size;
  const float* x     = (const float*)d_in[0];
  const float* gamma = (const float*)d_in[1];
  const float* beta  = (const float*)d_in[2];
  const float* wqkv  = (const float*)d_in[3];
  const float* wout  = (const float*)d_in[4];
  float* out = (float*)d_out;

  char* ws = (char*)d_ws;
  _Float16* xn    = (_Float16*)(ws + 0);                      // 8192*1024*2 = 16,777,216
  _Float16* wqkvT = (_Float16*)(ws + 16777216);               // 1536*1024*2 =  3,145,728
  _Float16* woutT = (_Float16*)(ws + 19922944);               // 1024*512*2  =  1,048,576
  _Float16* qh    = (_Float16*)(ws + 20971520);               // 16*4096*64*2 = 8,388,608
  _Float16* kh    = (_Float16*)(ws + 29360128);
  _Float16* vth   = (_Float16*)(ws + 37748736);               // V transposed [b][h][dh][n]
  _Float16* ah    = (_Float16*)(ws + 46137344);               // 8192*512*2 = 8,388,608

  ln_f16_kernel<<<8192, 256, 0, stream>>>(x, gamma, beta, xn);
  convT_kernel<<<(1024 * 1536 + 255) / 256, 256, 0, stream>>>(wqkv, wqkvT, 1024, 1536);
  convT_kernel<<<(512 * 1024 + 255) / 256, 256, 0, stream>>>(wout, woutT, 512, 1024);
  gemm_qkv_kernel<<<dim3(24, 64), 256, 0, stream>>>(xn, wqkvT, qh, kh, vth);
  attn_kernel<<<dim3(64, 16), 128, 0, stream>>>(qh, kh, vth, ah);
  gemm_out_kernel<<<dim3(16, 64), 256, 0, stream>>>(ah, woutT, out);
}
